// FeatureGrid_22230750724639
// MI455X (gfx1250) — matrix operand order repaired and checked by reading, not compiled, not run on hardware
//
#include <hip/hip_runtime.h>
#include <hip/hip_fp16.h>

// CDNA5 / gfx1250: wave32, WMMA 16x16x32 f16 -> f32 accum, split-f16 for ~f32 accuracy.
typedef __attribute__((ext_vector_type(16))) _Float16 v16h;
typedef __attribute__((ext_vector_type(8)))  float    v8f;

#define B_  4
#define C_  64
#define H_  128
#define W_  128
#define N_  (H_*W_)      // 16384
#define M_  (B_*N_)      // 65536 tokens total
#define OC_ 89

__device__ __forceinline__ int iclamp(int v, int lo, int hi) {
    return v < lo ? lo : (v > hi ? hi : v);
}

struct Bil {
    int x0, x1, y0, y1;        // clipped corner indices
    float w00, w10, w01, w11;  // weights * validity (zero padding)
};

__device__ __forceinline__ Bil bil_setup(float x, float y, int W, int H) {
    float ixf = ((x + 1.f) * (float)W - 1.f) * 0.5f;
    float iyf = ((y + 1.f) * (float)H - 1.f) * 0.5f;
    float x0f = floorf(ixf), y0f = floorf(iyf);
    float wx1 = ixf - x0f, wy1 = iyf - y0f;
    float wx0 = 1.f - wx1, wy0 = 1.f - wy1;
    int x0 = (int)x0f, y0 = (int)y0f;
    float vx0 = (x0 >= 0 && x0 < W) ? 1.f : 0.f;
    float vx1 = (x0 + 1 >= 0 && x0 + 1 < W) ? 1.f : 0.f;
    float vy0 = (y0 >= 0 && y0 < H) ? 1.f : 0.f;
    float vy1 = (y0 + 1 >= 0 && y0 + 1 < H) ? 1.f : 0.f;
    Bil r;
    r.w00 = wx0 * wy0 * vx0 * vy0;
    r.w10 = wx1 * wy0 * vx1 * vy0;
    r.w01 = wx0 * wy1 * vx0 * vy1;
    r.w11 = wx1 * wy1 * vx1 * vy1;
    r.x0 = iclamp(x0, 0, W - 1);     r.x1 = iclamp(x0 + 1, 0, W - 1);
    r.y0 = iclamp(y0, 0, H - 1);     r.y1 = iclamp(y0 + 1, 0, H - 1);
    return r;
}

// ---------------------------------------------------------------------------
// Kernel 0: per-token bilinear sampling of latents (64ch), image (1ch),
// virtual fcoord grid (2ch) + positional encoding. Writes out channels 64..88
// and the raw q_features buffer (f32, [B,N,64]).
// ---------------------------------------------------------------------------
__global__ void __launch_bounds__(256)
k_sample(const float* __restrict__ image,    // [B,1,256,256]
         const float* __restrict__ latents,  // [B,64,128,128]
         const float* __restrict__ coords,   // [B,N,2]  (y,x)
         float* __restrict__ qf,             // [B,N,64]
         float* __restrict__ out)            // [B,N,89]
{
    int t = blockIdx.x * blockDim.x + threadIdx.x;
    if (t >= M_) return;
    int b = t / N_;
    float yc = coords[(size_t)t * 2 + 0];
    float xc = coords[(size_t)t * 2 + 1];   // xy = flip(coords)

    // --- latents (H=W=128, 64 channels) ---
    {
        Bil bl = bil_setup(xc, yc, W_, H_);
        const float* Lb = latents + (size_t)b * C_ * N_;
        size_t i00 = (size_t)bl.y0 * W_ + bl.x0;
        size_t i10 = (size_t)bl.y0 * W_ + bl.x1;
        size_t i01 = (size_t)bl.y1 * W_ + bl.x0;
        size_t i11 = (size_t)bl.y1 * W_ + bl.x1;
        float* qrow = qf + (size_t)t * C_;
        #pragma unroll 4
        for (int c = 0; c < C_; ++c) {
            const float* p = Lb + (size_t)c * N_;
            qrow[c] = bl.w00 * p[i00] + bl.w10 * p[i10] +
                      bl.w01 * p[i01] + bl.w11 * p[i11];
        }
    }

    float* orow = out + (size_t)t * OC_;

    // --- image (256x256, 1 channel) ---
    {
        Bil bl = bil_setup(xc, yc, 256, 256);
        const float* Ib = image + (size_t)b * 256 * 256;
        float v = bl.w00 * Ib[(size_t)bl.y0 * 256 + bl.x0] +
                  bl.w10 * Ib[(size_t)bl.y0 * 256 + bl.x1] +
                  bl.w01 * Ib[(size_t)bl.y1 * 256 + bl.x0] +
                  bl.w11 * Ib[(size_t)bl.y1 * 256 + bl.x1];
        orow[64] = v;
    }

    // --- fcoord (virtual linear ramps, 128x128, 2 channels) + pos enc ---
    {
        Bil bl = bil_setup(xc, yc, W_, H_);
        float fy0 = -1.f + (2.f * bl.y0 + 1.f) / (float)H_;
        float fy1 = -1.f + (2.f * bl.y1 + 1.f) / (float)H_;
        float fx0 = -1.f + (2.f * bl.x0 + 1.f) / (float)W_;
        float fx1 = -1.f + (2.f * bl.x1 + 1.f) / (float)W_;
        // channel 0: value = fy[row]; channel 1: value = fx[col]
        float qc0 = bl.w00 * fy0 + bl.w10 * fy0 + bl.w01 * fy1 + bl.w11 * fy1;
        float qc1 = bl.w00 * fx0 + bl.w10 * fx1 + bl.w01 * fx0 + bl.w11 * fx1;
        float c0 = (qc0 + 1.f) * 0.5f;
        float c1 = (qc1 + 1.f) * 0.5f;
        const float PI = 3.14159265358979323846f;
        #pragma unroll
        for (int o = 0; o < 6; ++o) {
            float f = PI * (float)(1 << o);
            // accurate libm sin/cos (reference compares PE channels directly)
            orow[65 + 0 + o]  = sinf(c0 * f);   // sin, ch0
            orow[65 + 6 + o]  = sinf(c1 * f);   // sin, ch1
            orow[65 + 12 + o] = cosf(c0 * f);   // cos, ch0
            orow[65 + 18 + o] = cosf(c1 * f);   // cos, ch1
        }
    }
}

// ---------------------------------------------------------------------------
// Kernel 1: LayerNorm. Per token t=(b,p):
//   xq[t,:]  = LN_q(qf[t,:])                    (contiguous row)
//   cn[t,:]  = LN_c(uf[b,:,h,w])  where uf is the flat reinterpretation:
//              uf[b,c,p] = qf_flat[b*N*64 + c*N + p]   (stride-N gather)
// f32 outputs (GEMM splits to f16 hi/lo on the fly).
// ---------------------------------------------------------------------------
__global__ void __launch_bounds__(256)
k_ln(const float* __restrict__ qf,
     const float* __restrict__ lnq_s, const float* __restrict__ lnq_b,
     const float* __restrict__ lnc_s, const float* __restrict__ lnc_b,
     float* __restrict__ xq, float* __restrict__ cn)
{
    int t = blockIdx.x * blockDim.x + threadIdx.x;
    if (t >= M_) return;
    int b = t / N_;
    int p = t - b * N_;

    // (a) token-row LN -> xq
    {
        const float4* r4 = (const float4*)(qf + (size_t)t * C_);
        float v[C_];
        #pragma unroll
        for (int i = 0; i < C_ / 4; ++i) {
            float4 q = r4[i];
            v[4*i] = q.x; v[4*i+1] = q.y; v[4*i+2] = q.z; v[4*i+3] = q.w;
        }
        float mu = 0.f;
        #pragma unroll
        for (int c = 0; c < C_; ++c) mu += v[c];
        mu *= (1.f / C_);
        float var = 0.f;
        #pragma unroll
        for (int c = 0; c < C_; ++c) { float d = v[c] - mu; var += d * d; }
        var *= (1.f / C_);
        float inv = rsqrtf(var + 1e-5f);
        float* xr = xq + (size_t)t * C_;
        #pragma unroll
        for (int c = 0; c < C_; ++c)
            xr[c] = (v[c] - mu) * inv * lnq_s[c] + lnq_b[c];
    }

    // (b) grid-column LN -> cn
    {
        const float* base = qf + (size_t)b * N_ * C_ + p;
        float v[C_];
        #pragma unroll 8
        for (int c = 0; c < C_; ++c) v[c] = base[(size_t)c * N_];
        float mu = 0.f;
        #pragma unroll
        for (int c = 0; c < C_; ++c) mu += v[c];
        mu *= (1.f / C_);
        float var = 0.f;
        #pragma unroll
        for (int c = 0; c < C_; ++c) { float d = v[c] - mu; var += d * d; }
        var *= (1.f / C_);
        float inv = rsqrtf(var + 1e-5f);
        float* cr = cn + (size_t)t * C_;
        #pragma unroll
        for (int c = 0; c < C_; ++c)
            cr[c] = (v[c] - mu) * inv * lnc_s[c] + lnc_b[c];
    }
}

// ---------------------------------------------------------------------------
// Kernel 2: split-f16 GEMM  Cmat[M,Nc] = A[M,64](f32) @ Bw[64,Nc](f32) (+bias)
// x = hi + lo (both f16);  A*B ~= Ah*Bh + Ah*Bl + Al*Bh  (3 WMMAs / k-step,
// ~1e-6 rel error).  One wave owns 4 consecutive 16-row tiles of one 16-col
// stripe, so split B fragments (both k-steps) are loaded once, reused 4x.
// Documented wave32 VGPR layouts (CDNA5 ISA §7.12.2):
//   A: m = lane&15; VGPR j holds K = 2j + (j>=4?8:0) + (lane>=16?8:0), +1
//   B: n = lane&15; VGPR j holds K = 2j + (j>=4?8:0) + (lane>=16?8:0), +1
//   C: N = lane&15; VGPR r holds M = r + (lane>=16?8:0)
// ---------------------------------------------------------------------------
template<bool HAS_BIAS>
__global__ void __launch_bounds__(256)
k_gemm(const float* __restrict__ A,     // [M,64] row-major f32
       const float* __restrict__ Bw,    // [64,Nc] row-major f32
       const float* __restrict__ bias,  // [Nc] (used iff HAS_BIAS)
       float* __restrict__ Cmat,        // row stride ldc
       int Nc, int ldc, int nGroups)
{
    int lane = threadIdx.x & 31;
    int wave = threadIdx.x >> 5;
    int grp  = blockIdx.x * (blockDim.x >> 5) + wave;
    if (grp >= nGroups) return;          // wave-uniform (EXEC all-1 for WMMA)
    int tiles_n = Nc >> 4;
    int gm = grp / tiles_n;              // group of 4 M-tiles
    int tn = grp - gm * tiles_n;
    int n0 = tn << 4;
    int hi = lane >> 4;                  // lane group
    int ln = lane & 15;

    // Load + split B fragments for both k-steps; reused across 4 M-tiles.
    v16h bh[2], blo[2];
    #pragma unroll
    for (int ks = 0; ks < 2; ++ks) {
        int k0 = ks * 32;
        #pragma unroll
        for (int j = 0; j < 8; ++j) {
            int Kb = 2 * j + (j >= 4 ? 8 : 0) + hi * 8;
            float x0 = Bw[(size_t)(k0 + Kb) * Nc + n0 + ln];
            float x1 = Bw[(size_t)(k0 + Kb + 1) * Nc + n0 + ln];
            _Float16 h0 = (_Float16)x0, h1 = (_Float16)x1;
            bh[ks][2*j]   = h0;
            bh[ks][2*j+1] = h1;
            blo[ks][2*j]   = (_Float16)(x0 - (float)h0);
            blo[ks][2*j+1] = (_Float16)(x1 - (float)h1);
        }
    }
    float bcol = HAS_BIAS ? bias[n0 + ln] : 0.f;

    #pragma unroll
    for (int mi = 0; mi < 4; ++mi) {
        int m0 = (gm * 4 + mi) << 4;
        const float* Arow = A + (size_t)(m0 + ln) * 64;
        v8f acc = {};
        #pragma unroll
        for (int ks = 0; ks < 2; ++ks) {
            int k0 = ks * 32;
            v16h ah, al;
            #pragma unroll
            for (int j = 0; j < 8; ++j) {
                int Ka = 2 * j + (j >= 4 ? 8 : 0) + hi * 8;
                float x0 = Arow[k0 + Ka];
                float x1 = Arow[k0 + Ka + 1];
                _Float16 h0 = (_Float16)x0, h1 = (_Float16)x1;
                ah[2*j]   = h0;
                ah[2*j+1] = h1;
                al[2*j]   = (_Float16)(x0 - (float)h0);
                al[2*j+1] = (_Float16)(x1 - (float)h1);
            }
            // small terms first, big term last
            acc = __builtin_amdgcn_wmma_f32_16x16x32_f16(
                      false, al, false, bh[ks], (short)0, acc, false, false);
            acc = __builtin_amdgcn_wmma_f32_16x16x32_f16(
                      false, ah, false, blo[ks], (short)0, acc, false, false);
            acc = __builtin_amdgcn_wmma_f32_16x16x32_f16(
                      false, ah, false, bh[ks], (short)0, acc, false, false);
        }
        #pragma unroll
        for (int r = 0; r < 8; ++r) {
            size_t row = (size_t)(m0 + r + hi * 8);
            Cmat[row * (size_t)ldc + n0 + ln] = acc[r] + bcol;
        }
    }
}

// ---------------------------------------------------------------------------
// Kernel 3: 9-neighbor attention per token. KV rows are shared per grid
// location (computed once), gathered with edge clipping.
// ---------------------------------------------------------------------------
__global__ void __launch_bounds__(256)
k_attn(const float* __restrict__ qbuf,  // [M,64]
       const float* __restrict__ kvb,   // [M,128] (k | v)
       float* __restrict__ obuf)        // [M,64]
{
    int t = blockIdx.x * blockDim.x + threadIdx.x;
    if (t >= M_) return;
    int b = t / N_;
    int p = t - b * N_;
    int h = p >> 7;
    int w = p & (W_ - 1);

    float q[C_];
    {
        const float4* q4 = (const float4*)(qbuf + (size_t)t * C_);
        #pragma unroll
        for (int i = 0; i < C_ / 4; ++i) {
            float4 v = q4[i];
            q[4*i] = v.x; q[4*i+1] = v.y; q[4*i+2] = v.z; q[4*i+3] = v.w;
        }
    }

    int rows[9];
    float sim[9];
    int kk = 0;
    #pragma unroll
    for (int di = -1; di <= 1; ++di) {
        #pragma unroll
        for (int dj = -1; dj <= 1; ++dj) {
            int hh = iclamp(h + di, 0, H_ - 1);
            int ww = iclamp(w + dj, 0, W_ - 1);
            int r = b * N_ + hh * W_ + ww;
            rows[kk] = r;
            const float4* k4 = (const float4*)(kvb + (size_t)r * 128);
            float s = 0.f;
            #pragma unroll
            for (int i = 0; i < C_ / 4; ++i) {
                float4 v = k4[i];
                s += q[4*i] * v.x + q[4*i+1] * v.y + q[4*i+2] * v.z + q[4*i+3] * v.w;
            }
            sim[kk] = s * 0.125f;   // * C^-0.5
            ++kk;
        }
    }

    float mx = sim[0];
    #pragma unroll
    for (int i = 1; i < 9; ++i) mx = fmaxf(mx, sim[i]);
    float es[9], sum = 0.f;
    #pragma unroll
    for (int i = 0; i < 9; ++i) { es[i] = expf(sim[i] - mx); sum += es[i]; }
    float rs = 1.f / sum;

    float o[C_];
    #pragma unroll
    for (int c = 0; c < C_; ++c) o[c] = 0.f;
    #pragma unroll
    for (int i = 0; i < 9; ++i) {
        float a = es[i] * rs;
        const float4* v4 = (const float4*)(kvb + (size_t)rows[i] * 128 + 64);
        #pragma unroll
        for (int j = 0; j < C_ / 4; ++j) {
            float4 v = v4[j];
            o[4*j]   += a * v.x;  o[4*j+1] += a * v.y;
            o[4*j+2] += a * v.z;  o[4*j+3] += a * v.w;
        }
    }
    float* orow = obuf + (size_t)t * C_;
    #pragma unroll
    for (int c = 0; c < C_; ++c) orow[c] = o[c];
}

// ---------------------------------------------------------------------------
extern "C" void kernel_launch(void* const* d_in, const int* in_sizes, int n_in,
                              void* d_out, int out_size, void* d_ws, size_t ws_size,
                              hipStream_t stream)
{
    const float* image   = (const float*)d_in[0];
    const float* latents = (const float*)d_in[1];
    const float* coords  = (const float*)d_in[2];
    const float* lnq_s   = (const float*)d_in[3];
    const float* lnq_b   = (const float*)d_in[4];
    const float* lnc_s   = (const float*)d_in[5];
    const float* lnc_b   = (const float*)d_in[6];
    const float* w_q     = (const float*)d_in[7];
    const float* w_kv    = (const float*)d_in[8];
    const float* w_out   = (const float*)d_in[9];
    const float* b_out   = (const float*)d_in[10];
    float* out = (float*)d_out;

    // Workspace layout with buffer aliasing (lifetimes are disjoint in stream
    // order): qf dead after k_ln -> reused as qb; xq dead after Q-GEMM ->
    // reused as o. Total 80 MB, fully L2-resident on MI455X (192 MB).
    char* ws = (char*)d_ws;
    size_t off = 0;
    float* qf  = (float*)(ws + off);  off += (size_t)M_ * C_ * 4;   // 16 MB (later qb)
    float* xq  = (float*)(ws + off);  off += (size_t)M_ * C_ * 4;   // 16 MB (later o)
    float* cn  = (float*)(ws + off);  off += (size_t)M_ * C_ * 4;   // 16 MB
    float* kvb = (float*)(ws + off);  off += (size_t)M_ * 128 * 4;  // 32 MB
    float* qb  = qf;   // alias: qf consumed by k_ln before Q-GEMM writes qb
    float* ob  = xq;   // alias: xq consumed by Q-GEMM before k_attn writes o

    dim3 blk(256);

    k_sample<<<M_ / 256, blk, 0, stream>>>(image, latents, coords, qf, out);
    k_ln<<<M_ / 256, blk, 0, stream>>>(qf, lnq_s, lnq_b, lnc_s, lnc_b, xq, cn);

    {   // Q = xq @ w_q   [65536 x 64]
        int nGroups = (M_ / 64) * (64 / 16);      // 4096
        k_gemm<false><<<nGroups / 8, blk, 0, stream>>>(xq, w_q, nullptr, qb, 64, 64, nGroups);
    }
    {   // KV = cn @ w_kv [65536 x 128]
        int nGroups = (M_ / 64) * (128 / 16);     // 8192
        k_gemm<false><<<nGroups / 8, blk, 0, stream>>>(cn, w_kv, nullptr, kvb, 128, 128, nGroups);
    }

    k_attn<<<M_ / 256, blk, 0, stream>>>(qb, kvb, ob);

    {   // out[:, 0:64] = o @ w_out + b_out   (row stride 89 into d_out)
        int nGroups = (M_ / 64) * (64 / 16);      // 4096
        k_gemm<true><<<nGroups / 8, blk, 0, stream>>>(ob, w_out, b_out, out, 64, OC_, nGroups);
    }
}
